// AngleFusion_31499290149278
// MI455X (gfx1250) — matrix operand order repaired and checked by reading, not compiled, not run on hardware
//
#include <hip/hip_runtime.h>
#include <hip/hip_bf16.h>

// ---------------------------------------------------------------------------
// MI455X (gfx1250) implementation.
//   B=32, C=512, H=W=32, LEN=1024, NH=2, N_CH=C*NH=1024
// bf16 V_WMMA_F32_16X16X32_BF16 for all three big contractions, fp32 rest.
// B-operands pre-swizzled so each WMMA fragment load is two contiguous b128
// loads per lane. Intermediates stay resident in the 192MB L2.
// Attention kernel uses GLOBAL_LOAD_ASYNC_TO_LDS_B128 (ASYNCcnt) when the
// toolchain exposes it.
// ---------------------------------------------------------------------------

typedef unsigned short u16;
typedef __attribute__((ext_vector_type(4)))  int             v4i;
typedef __attribute__((ext_vector_type(16))) __bf16          v16bf;
typedef __attribute__((ext_vector_type(16))) unsigned short  v16us;
typedef __attribute__((ext_vector_type(8)))  float           v8f;

#ifndef __has_builtin
#define __has_builtin(x) 0
#endif
#if __has_builtin(__builtin_amdgcn_global_load_async_to_lds_b128) && \
    __has_builtin(__builtin_amdgcn_s_wait_asynccnt)
#define HAVE_ASYNC_LDS 1
#else
#define HAVE_ASYNC_LDS 0
#endif

#if HAVE_ASYNC_LDS
typedef __attribute__((address_space(1))) v4i* gv4p;   // global int4*
typedef __attribute__((address_space(3))) v4i* lv4p;   // LDS int4*
static __device__ __forceinline__ void async_copy16(void* lds, const void* g) {
    __builtin_amdgcn_global_load_async_to_lds_b128(
        (gv4p)(v4i*)(void*)g, (lv4p)(v4i*)lds, 0, 0);
}
#endif

union FragU {
    uint4  q[2];   // two b128 loads
    v16us  us;     // per-element bit fill
    v16bf  bf;     // WMMA operand view
};

static __device__ __forceinline__ u16 f2bf(float f) {
    unsigned int u = __float_as_uint(f);
    unsigned int lsb = (u >> 16) & 1u;
    u += 0x7fffu + lsb;              // round-to-nearest-even
    return (u16)(u >> 16);
}

// B-operand (16-bit, 32x16) per-lane K index, e = 2r + p: K = hi*8 + (e&7) + (e>>3)*16
static __device__ __forceinline__ int b_kidx(int r, int p, int hi) {
    return hi * 8 + ((2 * r + p) & 7) + ((2 * r + p) >> 3) * 16;
}

// ---------------------------------------------------------------------------
// fp32 -> bf16 copy (featuremap, conv_w)
// ---------------------------------------------------------------------------
__global__ void k_cvt(const float* __restrict__ src, u16* __restrict__ dst, int n) {
    for (int i = blockIdx.x * blockDim.x + threadIdx.x; i < n;
         i += gridDim.x * blockDim.x)
        dst[i] = f2bf(src[i]);
}

// ---------------------------------------------------------------------------
// wmh [K=1024][N=2048] fp32 -> B-fragment-swizzled bf16:
// layout [nt(128)][kt(32)][lane(32)][16], element e=(r*2+p) holds
// wmh[kt*32 + hi*8 + (e&7) + (e>>3)*16][nt*16 + (lane&15)]
// ---------------------------------------------------------------------------
__global__ void k_swz_wmh(const float* __restrict__ wmh, u16* __restrict__ dst) {
    int idx = blockIdx.x * blockDim.x + threadIdx.x;    // 2,097,152 total
    if (idx >= 1024 * 2048) return;
    int e    = idx & 15;
    int lane = (idx >> 4) & 31;
    int kt   = (idx >> 9) & 31;
    int nt   = idx >> 14;
    int r = e >> 1, p = e & 1, hi = lane >> 4, col = lane & 15;
    int k = kt * 32 + b_kidx(r, p, hi);
    int n = nt * 16 + col;
    dst[idx] = f2bf(wmh[(size_t)k * 2048 + n]);
}

// ---------------------------------------------------------------------------
// angle MLP: 1 -> 256 -> 512 -> 1024 (ReLU each); one block per batch.
// Output amap bf16 [B][1024] (amap[w][h] row-major == av linear).
// ---------------------------------------------------------------------------
__global__ void k_mlp(const float* __restrict__ angle,
                      const float* __restrict__ w1, const float* __restrict__ b1,
                      const float* __restrict__ w2, const float* __restrict__ b2,
                      const float* __restrict__ w3, const float* __restrict__ b3,
                      u16* __restrict__ amap_bf) {
    __shared__ float l1[256];
    __shared__ float l2[512];
    const int t = threadIdx.x, b = blockIdx.x;
    const float a = angle[b];
    l1[t] = fmaxf(a * w1[t] + b1[t], 0.f);
    __syncthreads();
    #pragma unroll 1
    for (int jj = 0; jj < 2; ++jj) {
        int j = t + jj * 256;
        float s = b2[j];
        for (int i = 0; i < 256; ++i) s += l1[i] * w2[i * 512 + j];
        l2[j] = fmaxf(s, 0.f);
    }
    __syncthreads();
    #pragma unroll 1
    for (int kk = 0; kk < 4; ++kk) {
        int k = t + kk * 256;
        float s = b3[k];
        for (int j = 0; j < 512; ++j) s += l2[j] * w3[j * 1024 + k];
        amap_bf[b * 1024 + k] = f2bf(fmaxf(s, 0.f));
    }
}

// ---------------------------------------------------------------------------
// fm GEMM: [16384,1024]bf16 @ wmh_swz -> fm bf16 [16384][2048] (+bmh).
// Block tile 128x128, 8 waves (4x2), wave tile 32x64 -> 2x4 accumulators.
// ---------------------------------------------------------------------------
__global__ void k_gemm_fm(const u16* __restrict__ A,        // fmap bf16 [16384][1024]
                          const u16* __restrict__ Bswz,     // wmh swizzled
                          const float* __restrict__ bmh,    // [2048]
                          u16* __restrict__ fm) {           // [16384][2048]
    const int tid = threadIdx.x;
    const int wv = tid >> 5, lane = tid & 31;
    const int colL = lane & 15, hi = lane >> 4;
    const int mW = blockIdx.x * 128 + (wv & 3) * 32;
    const int nW = blockIdx.y * 128 + (wv >> 2) * 64;

    v8f acc[2][4];
    #pragma unroll
    for (int mi = 0; mi < 2; ++mi)
        #pragma unroll
        for (int ni = 0; ni < 4; ++ni) acc[mi][ni] = (v8f)(0.f);

    #pragma unroll 1
    for (int kt = 0; kt < 32; ++kt) {
        FragU fa[2], fb[4];
        #pragma unroll
        for (int mi = 0; mi < 2; ++mi) {
            const u16* pa = A + (size_t)(mW + mi * 16 + colL) * 1024 + kt * 32 + hi * 8;
            fa[mi].q[0] = *(const uint4*)pa;
            fa[mi].q[1] = *(const uint4*)(pa + 16);
            __builtin_prefetch(pa + 32, 0, 1);   // speculative: OOB is dropped
        }
        #pragma unroll
        for (int ni = 0; ni < 4; ++ni) {
            const u16* pb = Bswz + (((size_t)(nW / 16 + ni) * 32 + kt) * 32 + lane) * 16;
            fb[ni].q[0] = *(const uint4*)pb;
            fb[ni].q[1] = *(const uint4*)(pb + 8);
        }
        #pragma unroll
        for (int mi = 0; mi < 2; ++mi)
            #pragma unroll
            for (int ni = 0; ni < 4; ++ni)
                acc[mi][ni] = __builtin_amdgcn_wmma_f32_16x16x32_bf16(
                    false, fa[mi].bf, false, fb[ni].bf, (short)0, acc[mi][ni],
                    false, false);
    }

    float bias[4];
    #pragma unroll
    for (int ni = 0; ni < 4; ++ni) bias[ni] = bmh[nW + ni * 16 + colL];
    #pragma unroll
    for (int mi = 0; mi < 2; ++mi)
        #pragma unroll
        for (int ni = 0; ni < 4; ++ni)
            #pragma unroll
            for (int r = 0; r < 8; ++r) {
                int row = mW + mi * 16 + r + hi * 8;
                int col = nW + ni * 16 + colL;
                fm[(size_t)row * 2048 + col] = f2bf(acc[mi][ni][r] + bias[ni]);
            }
}

// ---------------------------------------------------------------------------
// Attention: per (b,c) one wave handles both heads.
//   fus = softmax_w(amap @ fm_tile)/sqrt(32);  fusion = fm_tile @ fus
// Output fusion in B-fragment-swizzled bf16 [b][nt=hw/16][kt=n/32][lane][16].
// Block = 4 waves covering 4 consecutive c (c0 % 4 == 0) so the block can
// emit coalesced 16B swizzled runs. LDS fills go through
// GLOBAL_LOAD_ASYNC_TO_LDS_B128 when available.
// ---------------------------------------------------------------------------
__global__ void k_attn(const u16* __restrict__ amap_bf,   // [B][1024]
                       const u16* __restrict__ fm,        // [16384][2048]
                       u16* __restrict__ fus_swz) {
    __shared__ __align__(16) u16   amapS[1024];
    __shared__ __align__(16) u16   fmS[4][2048];   // [wave][head*1024 + h*32 + w]
    __shared__ float t32S[4][32 * 33];             // logits / exp scratch (padded)
    __shared__ __align__(16) u16   fusbS[4][1024]; // softmaxed B tile [w][v]
    __shared__ u16   fstage[8][1040];              // [cl*2+head][h*32+v] (padded)

    const int tid = threadIdx.x;
    const int wv = tid >> 5, lane = tid & 31;
    const int colL = lane & 15, hi = lane >> 4;
    const int b  = blockIdx.x >> 7;
    const int c0 = (blockIdx.x & 127) * 4;
    const int c  = c0 + wv;

#if HAVE_ASYNC_LDS
    {
        // amap: 2KB, one 16B async copy per thread (128 threads)
        async_copy16((char*)amapS + tid * 16,
                     (const char*)(amap_bf + b * 1024) + tid * 16);
        // fm tile: 4KB per wave, 8 x 16B per lane, direct to this wave's slice
        const char* src = (const char*)(fm + (size_t)(b * 512 + c) * 2048);
        char* dst = (char*)fmS[wv];
        #pragma unroll
        for (int j = 0; j < 8; ++j) {
            int off = (j * 32 + lane) * 16;
            async_copy16(dst + off, src + off);
        }
        __builtin_amdgcn_s_wait_asynccnt(0);
    }
#else
    for (int i = tid; i < 1024; i += 128) amapS[i] = amap_bf[b * 1024 + i];
    {
        const u16* src = fm + (size_t)(b * 512 + c) * 2048;
        for (int i = lane; i < 2048; i += 32) fmS[wv][i] = src[i];
    }
#endif
    __syncthreads();

    #pragma unroll 1
    for (int head = 0; head < 2; ++head) {
        const int hb = head * 1024;
        // ---- matmul1: fus = amap(A) @ fm(B), K = h = 32 ----
        // A fragment = two contiguous 16B runs per lane (K-major layout)
        FragU a1[2], b1[2];
        #pragma unroll
        for (int mi = 0; mi < 2; ++mi) {
            const u16* p = &amapS[(mi * 16 + colL) * 32 + hi * 8];
            a1[mi].q[0] = *(const uint4*)p;
            a1[mi].q[1] = *(const uint4*)(p + 16);
        }
        #pragma unroll
        for (int ni = 0; ni < 2; ++ni)
            #pragma unroll
            for (int r = 0; r < 8; ++r)
                #pragma unroll
                for (int p = 0; p < 2; ++p)
                    b1[ni].us[r * 2 + p] =
                        fmS[wv][hb + b_kidx(r, p, hi) * 32 + ni * 16 + colL];
        v8f cc[2][2];
        #pragma unroll
        for (int mi = 0; mi < 2; ++mi)
            #pragma unroll
            for (int ni = 0; ni < 2; ++ni)
                cc[mi][ni] = __builtin_amdgcn_wmma_f32_16x16x32_bf16(
                    false, a1[mi].bf, false, b1[ni].bf, (short)0, (v8f)(0.f),
                    false, false);
        // dump logits [w][v] to LDS
        #pragma unroll
        for (int mi = 0; mi < 2; ++mi)
            #pragma unroll
            for (int ni = 0; ni < 2; ++ni)
                #pragma unroll
                for (int r = 0; r < 8; ++r)
                    t32S[wv][(mi * 16 + r + hi * 8) * 33 + ni * 16 + colL] =
                        cc[mi][ni][r];
        __syncthreads();

        // ---- softmax over w (rows) per column v = lane, then /sqrt(32) ----
        {
            float m = -3.4e38f;
            for (int w = 0; w < 32; ++w) m = fmaxf(m, t32S[wv][w * 33 + lane]);
            float s = 0.f;
            for (int w = 0; w < 32; ++w) {
                float e = __expf(t32S[wv][w * 33 + lane] - m);
                s += e;
                t32S[wv][w * 33 + lane] = e;
            }
            float inv = 1.0f / (s * 5.65685424949238f);   // * sqrt(32)
            for (int w = 0; w < 32; ++w)
                fusbS[wv][w * 32 + lane] = f2bf(t32S[wv][w * 33 + lane] * inv);
        }
        __syncthreads();

        // ---- matmul2: fusion = fm(A, M=h K=w) @ fus(B), K = w = 32 ----
        FragU a2[2], b2[2];
        #pragma unroll
        for (int mi = 0; mi < 2; ++mi) {
            const u16* p = &fmS[wv][hb + (mi * 16 + colL) * 32 + hi * 8];
            a2[mi].q[0] = *(const uint4*)p;
            a2[mi].q[1] = *(const uint4*)(p + 16);
        }
        #pragma unroll
        for (int ni = 0; ni < 2; ++ni)
            #pragma unroll
            for (int r = 0; r < 8; ++r)
                #pragma unroll
                for (int p = 0; p < 2; ++p)
                    b2[ni].us[r * 2 + p] =
                        fusbS[wv][b_kidx(r, p, hi) * 32 + ni * 16 + colL];
        v8f dd[2][2];
        #pragma unroll
        for (int mi = 0; mi < 2; ++mi)
            #pragma unroll
            for (int ni = 0; ni < 2; ++ni)
                dd[mi][ni] = __builtin_amdgcn_wmma_f32_16x16x32_bf16(
                    false, a2[mi].bf, false, b2[ni].bf, (short)0, (v8f)(0.f),
                    false, false);
        // stage fusion tile bf16: plane = wv*2 + head
        #pragma unroll
        for (int mi = 0; mi < 2; ++mi)
            #pragma unroll
            for (int ni = 0; ni < 2; ++ni)
                #pragma unroll
                for (int r = 0; r < 8; ++r)
                    fstage[wv * 2 + head]
                          [(mi * 16 + r + hi * 8) * 32 + ni * 16 + colL] =
                        f2bf(dd[mi][ni][r]);
        __syncthreads();
    }

    // ---- cooperative swizzled store: block's 8 k-values (k=2c+head) form a
    //      contiguous 8-element (16B) run at offset o0 within each 16-run ----
    const int k0 = 2 * c0;
    const int kt = k0 >> 5;
    const int hi0 = (k0 >> 3) & 1;
    const int o0 = ((k0 >> 4) & 1) * 8;          // 0 or 8
    for (int hw = tid; hw < 1024; hw += 128) {
        u16 v[8];
        #pragma unroll
        for (int o = 0; o < 8; ++o) v[o] = fstage[o][hw];   // plane o == run off
        uint4 q;
        q.x = (unsigned)v[0] | ((unsigned)v[1] << 16);
        q.y = (unsigned)v[2] | ((unsigned)v[3] << 16);
        q.z = (unsigned)v[4] | ((unsigned)v[5] << 16);
        q.w = (unsigned)v[6] | ((unsigned)v[7] << 16);
        size_t eidx =
            ((((size_t)b * 64 + (hw >> 4)) * 32 + kt) * 32 +
             (hi0 * 16 + (hw & 15))) * 16 + o0;
        *(uint4*)(fus_swz + eidx) = q;
    }
}

// ---------------------------------------------------------------------------
// out GEMM per batch: conv_w[512,1024] @ fusion_swz_b -> fused epilogue:
// d_out = featuremap + gamma*(acc + conv_b)
// ---------------------------------------------------------------------------
__global__ void k_gemm_out(const u16* __restrict__ A,      // conv_w bf16 [512][1024]
                           const u16* __restrict__ Bswz,   // fusion swizzled
                           const float* __restrict__ conv_b,
                           const float* __restrict__ gamma,
                           const float* __restrict__ fmap,
                           float* __restrict__ out) {
    const int tid = threadIdx.x;
    const int wv = tid >> 5, lane = tid & 31;
    const int colL = lane & 15, hi = lane >> 4;
    const int b  = blockIdx.z;
    const int mW = blockIdx.x * 128 + (wv & 3) * 32;   // c
    const int nW = blockIdx.y * 128 + (wv >> 2) * 64;  // hw
    const u16* Bb = Bswz + (size_t)b * (64 * 32 * 32 * 16);

    v8f acc[2][4];
    #pragma unroll
    for (int mi = 0; mi < 2; ++mi)
        #pragma unroll
        for (int ni = 0; ni < 4; ++ni) acc[mi][ni] = (v8f)(0.f);

    #pragma unroll 1
    for (int kt = 0; kt < 32; ++kt) {
        FragU fa[2], fb[4];
        #pragma unroll
        for (int mi = 0; mi < 2; ++mi) {
            const u16* pa = A + (size_t)(mW + mi * 16 + colL) * 1024 + kt * 32 + hi * 8;
            fa[mi].q[0] = *(const uint4*)pa;
            fa[mi].q[1] = *(const uint4*)(pa + 16);
        }
        #pragma unroll
        for (int ni = 0; ni < 4; ++ni) {
            const u16* pb = Bb + (((size_t)(nW / 16 + ni) * 32 + kt) * 32 + lane) * 16;
            fb[ni].q[0] = *(const uint4*)pb;
            fb[ni].q[1] = *(const uint4*)(pb + 8);
            __builtin_prefetch(pb + 512, 0, 1);  // next kt (speculative)
        }
        #pragma unroll
        for (int mi = 0; mi < 2; ++mi)
            #pragma unroll
            for (int ni = 0; ni < 4; ++ni)
                acc[mi][ni] = __builtin_amdgcn_wmma_f32_16x16x32_bf16(
                    false, fa[mi].bf, false, fb[ni].bf, (short)0, acc[mi][ni],
                    false, false);
    }

    const float g = gamma[0];
    #pragma unroll
    for (int mi = 0; mi < 2; ++mi)
        #pragma unroll
        for (int ni = 0; ni < 4; ++ni)
            #pragma unroll
            for (int r = 0; r < 8; ++r) {
                int cc = mW + mi * 16 + r + hi * 8;
                int hw = nW + ni * 16 + colL;
                size_t idx = ((size_t)b * 512 + cc) * 1024 + hw;
                out[idx] = fmap[idx] + g * (acc[mi][ni][r] + conv_b[cc]);
            }
}

// ---------------------------------------------------------------------------
extern "C" void kernel_launch(void* const* d_in, const int* in_sizes, int n_in,
                              void* d_out, int out_size, void* d_ws, size_t ws_size,
                              hipStream_t stream) {
    (void)in_sizes; (void)n_in; (void)out_size; (void)ws_size;
    const float* fmap   = (const float*)d_in[0];
    const float* angle  = (const float*)d_in[1];
    const float* w1     = (const float*)d_in[2];
    const float* b1     = (const float*)d_in[3];
    const float* w2     = (const float*)d_in[4];
    const float* b2     = (const float*)d_in[5];
    const float* w3     = (const float*)d_in[6];
    const float* b3     = (const float*)d_in[7];
    const float* wmh    = (const float*)d_in[8];
    const float* bmh    = (const float*)d_in[9];
    const float* conv_w = (const float*)d_in[10];
    const float* conv_b = (const float*)d_in[11];
    const float* gamma  = (const float*)d_in[12];

    char* w = (char*)d_ws;
    u16* fmap_bf = (u16*)(w);                         //  33,554,432 B
    u16* wmh_swz = (u16*)(w + 33554432);              //   4,194,304 B
    u16* convw_bf= (u16*)(w + 37748736);              //   1,048,576 B
    u16* amap_bf = (u16*)(w + 38797312);              //      65,536 B
    u16* fm_bf   = (u16*)(w + 38862848);              //  67,108,864 B
    u16* fus_swz = (u16*)(w + 105971712);             //  67,108,864 B

    k_cvt<<<4096, 256, 0, stream>>>(fmap, fmap_bf, 32 * 512 * 1024);
    k_swz_wmh<<<8192, 256, 0, stream>>>(wmh, wmh_swz);
    k_cvt<<<2048, 256, 0, stream>>>(conv_w, convw_bf, 512 * 1024);
    k_mlp<<<32, 256, 0, stream>>>(angle, w1, b1, w2, b2, w3, b3, amap_bf);
    k_gemm_fm<<<dim3(128, 16), 256, 0, stream>>>(fmap_bf, wmh_swz, bmh, fm_bf);
    k_attn<<<4096, 128, 0, stream>>>(amap_bf, fm_bf, fus_swz);
    k_gemm_out<<<dim3(4, 8, 32), 256, 0, stream>>>(convw_bf, fus_swz, conv_b,
                                                   gamma, fmap, (float*)d_out);
}
